// CustomBlock_34711925686936
// MI455X (gfx1250) — matrix operand order repaired and checked by reading, not compiled, not run on hardware
//
#include <hip/hip_runtime.h>

typedef __attribute__((ext_vector_type(16))) _Float16 v16h;
typedef __attribute__((ext_vector_type(8)))  _Float16 v8h;
typedef __attribute__((ext_vector_type(8)))  float    v8f;

constexpr int Bn  = 2;
constexpr int Tn  = 1024;
constexpr int Cn  = 512;
constexpr int Hn  = 8;
constexpr int KVHn = 4;
constexpr int HDn = 64;
constexpr int D2n = 32;
constexpr float SCALEc = 0.125f;   // 1/sqrt(64)

__device__ __forceinline__ v8f wmma_f32_f16(v16h a, v16h b, v8f c) {
  // emits v_wmma_f32_16x16x32_f16
  return __builtin_amdgcn_wmma_f32_16x16x32_f16(false, a, false, b, (short)0, c, false, false);
}

__device__ __forceinline__ v16h pack2(v8h lo, v8h hi) {
  v16h r;
#pragma unroll
  for (int i = 0; i < 8; ++i) { r[i] = lo[i]; r[i + 8] = hi[i]; }
  return r;
}

// A-fragment (16x32 f16, M x K): lane m = ln, K halves {base..base+7} and {base+16..base+23}
__device__ __forceinline__ v16h pack_a16(const _Float16* p) {
  return pack2(*(const v8h*)p, *(const v8h*)(p + 16));
}

// B-fragment (K x N stored N-major): lane n = ln, K halves {base..base+7} and {base+16..base+23}
__device__ __forceinline__ v16h pack_b16_contig(const _Float16* p) {
  return pack2(*(const v8h*)p, *(const v8h*)(p + 16));
}

// ---------------------------------------------------------------- fp32 -> fp16
__global__ void k_cvt_f16(const float* __restrict__ s, _Float16* __restrict__ d, int n) {
  int i = blockIdx.x * blockDim.x + threadIdx.x;
  if (i < n) d[i] = (_Float16)s[i];
}

// fp32 (K x N, row-major) -> fp16 transposed (N x K): B-fragments become contiguous
__global__ void k_cvt_tr(const float* __restrict__ s, _Float16* __restrict__ d, int K, int N) {
  int idx = blockIdx.x * blockDim.x + threadIdx.x;
  if (idx < K * N) {
    const int n = idx / K, k = idx % K;         // contiguous writes
    d[idx] = (_Float16)s[(size_t)k * N + n];
  }
}

// ------------------------------------------------ QKV projection + RoPE + transpose
// grid.x = B*T/16 (row tiles), grid.y = H + 2*KVH (0..7:q, 8..11:k, 12..15:v), 128 thr
__global__ __launch_bounds__(128) void k_proj_rope(
    const _Float16* __restrict__ xh,
    const _Float16* __restrict__ WqT,   // (H*HD) x C
    const _Float16* __restrict__ WkT,   // (KVH*HD) x C
    const _Float16* __restrict__ WvT,   // (KVH*HD) x C
    const float* __restrict__ cosp, const float* __restrict__ sinp,
    _Float16* __restrict__ qT, _Float16* __restrict__ kT, float* __restrict__ vT)
{
  __shared__ float tile[16][64];
  const int lane = threadIdx.x & 31, wave = threadIdx.x >> 5;
  const int lg = lane >> 4, ln = lane & 15;
  const int g0 = blockIdx.x * 16;
  const int b = g0 / Tn, t0 = g0 % Tn;
  const int hh = blockIdx.y;

  const _Float16* Wt; int nbase;
  if (hh < Hn)             { Wt = WqT; nbase = hh * HDn; }
  else if (hh < Hn + KVHn) { Wt = WkT; nbase = (hh - Hn) * HDn; }
  else                     { Wt = WvT; nbase = (hh - Hn - KVHn) * HDn; }
  const int n0 = nbase + wave * 16;
  const _Float16* arow = xh + (size_t)(g0 + ln) * Cn;
  const _Float16* brow = Wt + (size_t)(n0 + ln) * Cn;

  v8f acc = {0,0,0,0,0,0,0,0};
  for (int kb = 0; kb < Cn / 32; ++kb) {
    v16h af = pack_a16(arow + kb * 32 + lg * 8);
    v16h bf = pack_b16_contig(brow + kb * 32 + lg * 8);
    acc = wmma_f32_f16(af, bf, acc);
  }
#pragma unroll
  for (int e = 0; e < 8; ++e) tile[e + lg * 8][wave * 16 + ln] = acc[e];
  __syncthreads();

  for (int idx = threadIdx.x; idx < 16 * 64; idx += 128) {
    const int r = idx >> 6, d = idx & 63;
    const int t = t0 + r;
    float val;
    if (hh < Hn + KVHn) {  // rope for q and k
      const int dd = d & 31;
      const float cv = cosp[t * D2n + dd], sv = sinp[t * D2n + dd];
      const float t1 = tile[r][dd], t2 = tile[r][dd + 32];
      val = (d < D2n) ? (t1 * cv + t2 * sv) : (t2 * cv - t1 * sv);
    } else {
      val = tile[r][d];
    }
    if (hh < Hn)
      qT[(((size_t)b * Hn + hh) * Tn + t) * HDn + d] = (_Float16)(val * SCALEc);
    else if (hh < Hn + KVHn)
      kT[(((size_t)b * KVHn + (hh - Hn)) * Tn + t) * HDn + d] = (_Float16)val;
    else
      vT[(((size_t)b * KVHn + (hh - Hn - KVHn)) * Tn + t) * HDn + d] = val;
  }
}

// ------------------------------------------------ per-row softmax stats (m, Z), online
// grid.x = T/16, grid.y = B*H, 128 thr (4 waves split column blocks)
__global__ __launch_bounds__(128) void k_stats(
    const _Float16* __restrict__ qT, const _Float16* __restrict__ kT,
    float2* __restrict__ stats)
{
  __shared__ float lm[4][16], lz[4][16];
  const int lane = threadIdx.x & 31, wave = threadIdx.x >> 5;
  const int lg = lane >> 4, ln = lane & 15;
  const int rt = blockIdx.x;
  const int bh = blockIdx.y;
  const int b = bh / Hn, h = bh % Hn;
  const int t0 = rt * 16;

  const _Float16* q = qT + ((size_t)bh * Tn + t0) * HDn;
  const _Float16* k = kT + ((size_t)b * KVHn + h / 2) * Tn * HDn;

  v16h aq0 = pack_a16(q + (size_t)ln * HDn + lg * 8);
  v16h aq1 = pack_a16(q + (size_t)ln * HDn + 32 + lg * 8);

  float mrun[8], zrun[8];
#pragma unroll
  for (int e = 0; e < 8; ++e) { mrun[e] = -1e30f; zrun[e] = 0.f; }

  for (int j = wave; j <= rt; j += 4) {
    const _Float16* kp = k + (size_t)(j * 16 + ln) * HDn + lg * 8;
    if (j + 4 <= rt)
      __builtin_prefetch((const void*)(kp + 4 * 16 * HDn), 0, 1);  // global_prefetch_b8
    v8f s = {0,0,0,0,0,0,0,0};
    s = wmma_f32_f16(aq0, pack_b16_contig(kp), s);
    s = wmma_f32_f16(aq1, pack_b16_contig(kp + 32), s);
#pragma unroll
    for (int e = 0; e < 8; ++e) {
      const int r = e + lg * 8;
      float sv = s[e];
      if (j == rt && ln > r) sv = -1e30f;   // causal mask on diagonal tile
      float tm = sv;
#pragma unroll
      for (int o = 1; o < 16; o <<= 1) tm = fmaxf(tm, __shfl_xor(tm, o, 16));
      const float nm = fmaxf(mrun[e], tm);
      float ex = __expf(sv - nm);
#pragma unroll
      for (int o = 1; o < 16; o <<= 1) ex += __shfl_xor(ex, o, 16);
      zrun[e] = zrun[e] * __expf(mrun[e] - nm) + ex;
      mrun[e] = nm;
    }
  }
  if (ln == 0) {
#pragma unroll
    for (int e = 0; e < 8; ++e) { lm[wave][e + lg * 8] = mrun[e]; lz[wave][e + lg * 8] = zrun[e]; }
  }
  __syncthreads();
  if (threadIdx.x < 16) {
    const int r = threadIdx.x;
    float m = lm[0][r];
    for (int w = 1; w < 4; ++w) m = fmaxf(m, lm[w][r]);
    float z = 0.f;
    for (int w = 0; w < 4; ++w) z += lz[w][r] * __expf(lm[w][r] - m);
    stats[(size_t)bh * Tn + t0 + r] = make_float2(m, z);
  }
}

// ------------------------------------------------ blocked forward substitution
// (A+I) y = v with A recomputed tile-by-tile; y kept LDS-resident, stored TRANSPOSED
// (y_tr[d][t]) so B-fragments are contiguous ds_load_b128s.
// grid = B*H workgroups, 256 thr (8 waves split off-diagonal j-pair blocks)
#define SWAVES 8
__global__ __launch_bounds__(256) void k_solve(
    const _Float16* __restrict__ qT, const _Float16* __restrict__ kT,
    const float* __restrict__ vT, const float2* __restrict__ stats,
    _Float16* __restrict__ yh)
{
  __shared__ _Float16 y_tr[HDn][Tn];            // 128 KB (of the 320 KB WGP LDS)
  __shared__ float accs[16][HDn];               // 4 KB
  __shared__ _Float16 awork[SWAVES][16][32];    // 8 KB: per-wave A-tile staging
  __shared__ float adiag[16][16];               // 1 KB
  __shared__ float sm[16], sz[16];

  const int lane = threadIdx.x & 31, wave = threadIdx.x >> 5;
  const int lg = lane >> 4, ln = lane & 15;
  const int bh = blockIdx.x;
  const int b = bh / Hn, h = bh % Hn;

  const _Float16* qb = qT + (size_t)bh * Tn * HDn;
  const _Float16* kbp = kT + ((size_t)b * KVHn + h / 2) * Tn * HDn;
  const float*    vb = vT + ((size_t)b * KVHn + h / 2) * Tn * HDn;
  const float2*   st = stats + (size_t)bh * Tn;

  for (int kb = 0; kb < Tn / 16; ++kb) {
    for (int idx = threadIdx.x; idx < 16 * HDn; idx += 256) ((float*)accs)[idx] = 0.f;
    if (threadIdx.x < 16) {
      float2 s = st[kb * 16 + threadIdx.x];
      sm[threadIdx.x] = s.x; sz[threadIdx.x] = s.y;
    }
    __syncthreads();

    v16h aq0 = pack_a16(qb + (size_t)(kb * 16 + ln) * HDn + lg * 8);
    v16h aq1 = pack_a16(qb + (size_t)(kb * 16 + ln) * HDn + 32 + lg * 8);

    v8f acc0 = {0,0,0,0,0,0,0,0}, acc1 = {0,0,0,0,0,0,0,0};
    v8f acc2 = {0,0,0,0,0,0,0,0}, acc3 = {0,0,0,0,0,0,0,0};

    // off-diagonal: acc += A[kb, 2p..2p+1] @ y[2p..2p+1]  (K = 32)
    for (int p = wave; 2 * p < kb; p += SWAVES) {
#pragma unroll
      for (int hf = 0; hf < 2; ++hf) {
        const int j = 2 * p + hf;
        if (j < kb) {
          const _Float16* kp = kbp + (size_t)(j * 16 + ln) * HDn + lg * 8;
          if (j + 2 * SWAVES < kb)
            __builtin_prefetch((const void*)(kp + 2 * SWAVES * 16 * HDn), 0, 1);
          v8f s = {0,0,0,0,0,0,0,0};
          s = wmma_f32_f16(aq0, pack_b16_contig(kp), s);
          s = wmma_f32_f16(aq1, pack_b16_contig(kp + 32), s);
#pragma unroll
          for (int e = 0; e < 8; ++e) {
            const int r = e + lg * 8;
            awork[wave][r][hf * 16 + ln] = (_Float16)(__expf(s[e] - sm[r]) / sz[r]);
          }
        } else {
#pragma unroll
          for (int e = 0; e < 8; ++e) awork[wave][e + lg * 8][hf * 16 + ln] = (_Float16)0.f;
        }
      }
      // re-read A tile as an A-fragment: two 16B ds_load_b128 per lane
      v16h af = pack2(*(const v8h*)&awork[wave][ln][lg * 8],
                      *(const v8h*)&awork[wave][ln][lg * 8 + 16]);
      const int kr = p * 32 + lg * 8;
      v16h b0 = pack_b16_contig(&y_tr[ln][kr]);
      v16h b1 = pack_b16_contig(&y_tr[16 + ln][kr]);
      v16h b2 = pack_b16_contig(&y_tr[32 + ln][kr]);
      v16h b3 = pack_b16_contig(&y_tr[48 + ln][kr]);
      acc0 = wmma_f32_f16(af, b0, acc0);
      acc1 = wmma_f32_f16(af, b1, acc1);
      acc2 = wmma_f32_f16(af, b2, acc2);
      acc3 = wmma_f32_f16(af, b3, acc3);
    }

    // diagonal A tile (masked), by last wave
    if (wave == SWAVES - 1) {
      v8f s = {0,0,0,0,0,0,0,0};
      const _Float16* kp = kbp + (size_t)(kb * 16 + ln) * HDn + lg * 8;
      s = wmma_f32_f16(aq0, pack_b16_contig(kp), s);
      s = wmma_f32_f16(aq1, pack_b16_contig(kp + 32), s);
#pragma unroll
      for (int e = 0; e < 8; ++e) {
        const int r = e + lg * 8;
        adiag[r][ln] = (ln <= r) ? (__expf(s[e] - sm[r]) / sz[r]) : 0.f;
      }
    }

    // deterministic ordered reduction of wave partials
    for (int w = 0; w < SWAVES; ++w) {
      if (wave == w) {
#pragma unroll
        for (int e = 0; e < 8; ++e) {
          const int r = e + lg * 8;
          accs[r][ln]      += acc0[e];
          accs[r][16 + ln] += acc1[e];
          accs[r][32 + ln] += acc2[e];
          accs[r][48 + ln] += acc3[e];
        }
      }
      __syncthreads();
    }

    // 16x16 triangular solve, one lane per column (64 lanes own one d each)
    if (threadIdx.x < HDn) {
      const int c = threadIdx.x;
      float ycol[16];
      for (int i = 0; i < 16; ++i) {
        float s = vb[(size_t)(kb * 16 + i) * HDn + c] - accs[i][c];
        for (int jj = 0; jj < i; ++jj) s -= adiag[i][jj] * ycol[jj];
        const float y = s / (1.0f + adiag[i][i]);
        ycol[i] = y;
        y_tr[c][kb * 16 + i] = (_Float16)y;
      }
    }
    __syncthreads();
  }

  // transposing export: yh is (bh, t, d) so the output projection reads contiguous A-frags
  for (int idx = threadIdx.x; idx < Tn * HDn; idx += 256) {
    const int t = idx >> 6, d = idx & 63;
    yh[(size_t)bh * Tn * HDn + idx] = y_tr[d][t];
  }
}

// ------------------------------------------------ out = x + y @ Wo
// grid.x = B*T/16, grid.y = C/64, 128 thr
__global__ __launch_bounds__(128) void k_outproj(
    const _Float16* __restrict__ yh, const _Float16* __restrict__ WoT,  // C x C (N x K)
    const float* __restrict__ x, float* __restrict__ out)
{
  const int lane = threadIdx.x & 31, wave = threadIdx.x >> 5;
  const int lg = lane >> 4, ln = lane & 15;
  const int g0 = blockIdx.x * 16;
  const int b = g0 / Tn, t0 = g0 % Tn;
  const int n0 = blockIdx.y * 64 + wave * 16;
  const _Float16* brow = WoT + (size_t)(n0 + ln) * Cn;

  v8f acc = {0,0,0,0,0,0,0,0};
  for (int kbv = 0; kbv < Cn / 32; ++kbv) {
    const int hh = (kbv * 32) >> 6;       // head for this K chunk (32 | 64, never crosses)
    const int d0 = (kbv * 32) & 63;
    v16h af = pack_a16(yh + (((size_t)b * Hn + hh) * Tn + (t0 + ln)) * HDn + d0 + lg * 8);
    v16h bf = pack_b16_contig(brow + kbv * 32 + lg * 8);
    acc = wmma_f32_f16(af, bf, acc);
  }
#pragma unroll
  for (int e = 0; e < 8; ++e) {
    const int r = e + lg * 8;
    const size_t o = (size_t)(g0 + r) * Cn + n0 + ln;
    out[o] = x[o] + acc[e];
  }
}

// ----------------------------------------------------------------------------
extern "C" void kernel_launch(void* const* d_in, const int* in_sizes, int n_in,
                              void* d_out, int out_size, void* d_ws, size_t ws_size,
                              hipStream_t stream) {
  (void)in_sizes; (void)n_in; (void)out_size; (void)ws_size;
  const float* x    = (const float*)d_in[0];
  const float* cosp = (const float*)d_in[1];
  const float* sinp = (const float*)d_in[2];
  const float* Wq   = (const float*)d_in[3];
  const float* Wk   = (const float*)d_in[4];
  const float* Wv   = (const float*)d_in[5];
  const float* Wo   = (const float*)d_in[6];
  float* out = (float*)d_out;

  char* p = (char*)d_ws;
  auto alloc = [&](size_t bytes) -> char* {
    char* r = p; p += (bytes + 255) & ~(size_t)255; return r;
  };
  _Float16* xh  = (_Float16*)alloc((size_t)Bn * Tn * Cn * 2);
  _Float16* WqT = (_Float16*)alloc((size_t)Cn * Hn * HDn * 2);
  _Float16* WkT = (_Float16*)alloc((size_t)Cn * KVHn * HDn * 2);
  _Float16* WvT = (_Float16*)alloc((size_t)Cn * KVHn * HDn * 2);
  _Float16* WoT = (_Float16*)alloc((size_t)Cn * Cn * 2);
  _Float16* qTp = (_Float16*)alloc((size_t)Bn * Hn * Tn * HDn * 2);
  _Float16* kTp = (_Float16*)alloc((size_t)Bn * KVHn * Tn * HDn * 2);
  float*    vTp = (float*)   alloc((size_t)Bn * KVHn * Tn * HDn * 4);
  float2*   stp = (float2*)  alloc((size_t)Bn * Hn * Tn * 8);
  _Float16* yhp = (_Float16*)alloc((size_t)Bn * Hn * Tn * HDn * 2);

  k_cvt_f16<<<(Bn * Tn * Cn + 255) / 256, 256, 0, stream>>>(x, xh, Bn * Tn * Cn);
  auto cvt_tr = [&](const float* s, _Float16* d, int K, int N) {
    k_cvt_tr<<<(K * N + 255) / 256, 256, 0, stream>>>(s, d, K, N);
  };
  cvt_tr(Wq, WqT, Cn, Hn * HDn);
  cvt_tr(Wk, WkT, Cn, KVHn * HDn);
  cvt_tr(Wv, WvT, Cn, KVHn * HDn);
  cvt_tr(Wo, WoT, Cn, Cn);

  k_proj_rope<<<dim3(Bn * Tn / 16, Hn + 2 * KVHn), 128, 0, stream>>>(
      xh, WqT, WkT, WvT, cosp, sinp, qTp, kTp, vTp);
  k_stats<<<dim3(Tn / 16, Bn * Hn), 128, 0, stream>>>(qTp, kTp, stp);
  k_solve<<<dim3(Bn * Hn), 256, 0, stream>>>(qTp, kTp, vTp, stp, yhp);
  k_outproj<<<dim3(Bn * Tn / 16, Cn / 64), 128, 0, stream>>>(yhp, WoT, x, out);
}
